// GRNLayer_9285719294119
// MI455X (gfx1250) — matrix operand order repaired and checked by reading, not compiled, not run on hardware
//
#include <hip/hip_runtime.h>

typedef __attribute__((ext_vector_type(16))) _Float16 v16h;
typedef __attribute__((ext_vector_type(8)))  _Float16 v8h;
typedef __attribute__((ext_vector_type(8)))  float    v8f;

#define NTOT   1024
#define NI     64
#define NO     64
#define BATCH  16384
#define WARMIT 25
#define NCOLS  960          // columns j = 64..1023
#define SBLK   30           // blocks per warm-step kernel (32 cols each)

// ---------------------------------------------------------------------------
// Kernel 1: Sfull[k][j] = mask(k) * (exp(-b|e_k-id_j|) - exp(-b|i_k-id_j|))
// ---------------------------------------------------------------------------
__global__ void grn_sig_kernel(const float* __restrict__ ids,
                               const float* __restrict__ enh,
                               const float* __restrict__ inh,
                               const float* __restrict__ beta,
                               float* __restrict__ Sfull) {
  int idx = blockIdx.x * blockDim.x + threadIdx.x;     // exactly 1024*1024
  int k = idx >> 10, j = idx & 1023;
  float b = beta[0];
  float e = __expf(-b * fabsf(enh[k] - ids[j]));
  float h = __expf(-b * fabsf(inh[k] - ids[j]));
  float m = (k >= NI && k < NI + NO) ? 0.f : 1.f;
  Sfull[idx] = m * (e - h);
}

// ---------------------------------------------------------------------------
// Init: raw conc buffer = 1/N
// ---------------------------------------------------------------------------
__global__ void grn_init_kernel(float* __restrict__ rawA) {
  rawA[threadIdx.x] = 1.0f / (float)NTOT;
}

// ---------------------------------------------------------------------------
// One warmup step, parallel over 30 blocks (no atomics: per-block partials,
// normalization applied lazily via norm = 1/sum of previous step's partials).
// True conc:  k<64 -> 1/N (never changes);  k>=64 -> rawIn[k]*norm.
// Each block: 32 columns x 8 k-chunks (256 threads).
// ---------------------------------------------------------------------------
__global__ void __launch_bounds__(256)
grn_step_kernel(const float* __restrict__ Sfull,
                const float* __restrict__ delta,
                const float* __restrict__ rawIn,
                float* __restrict__ rawOut,
                const float* __restrict__ prevPart, int prevCount,
                float* __restrict__ curPart) {
  __shared__ float pfix[256];   // partial of sum_{k<64} S[k][j]
  __shared__ float praw[256];   // partial of sum_{k>=64} rawIn[k]*S[k][j]
  __shared__ float red[32];

  const int t    = threadIdx.x;
  const int jloc = t & 31;
  const int kc   = t >> 5;                       // 0..7, 128 k each
  const int j    = NI + blockIdx.x * 32 + jloc;  // 64..1023
  const float invN   = 1.0f / (float)NTOT;
  const float dscale = delta[0] * invN;

  // s from previous iteration's per-block partials (fixed order -> determinism)
  float s = 0.f;
  for (int b = 0; b < prevCount; ++b) s += prevPart[b];
  const float norm = (s > 0.f) ? 1.0f / s : 1.0f;

  float pf = 0.f, pr = 0.f;
  const int k0 = kc * 128;
  for (int k = k0; k < k0 + 128; ++k) {
    float sv = Sfull[k * NTOT + j];
    if (k < NI) pf += sv;
    else        pr = fmaf(rawIn[k], sv, pr);
  }
  pfix[t] = pf;
  praw[t] = pr;
  __syncthreads();

  if (t < 32) {
    float df = 0.f, dr = 0.f;
    #pragma unroll
    for (int g = 0; g < 8; ++g) { df += pfix[t + 32 * g]; dr += praw[t + 32 * g]; }
    float dot = df * invN + dr * norm;
    float cj  = rawIn[j] * norm;
    float nj  = fmaxf(cj + dot * dscale, 0.f);
    rawOut[j] = nj;
    red[t] = nj;
  }
  __syncthreads();
  if (t == 0) {
    float bs = 0.f;
    #pragma unroll
    for (int i = 0; i < 32; ++i) bs += red[i];
    curPart[blockIdx.x] = bs;
  }
}

// ---------------------------------------------------------------------------
// Finalize: conc_warm[j] = rawLast[j]*norm;  base[j-64] = conc_warm[j] + v*d/N
// with v = sum_{k>=64} conc_warm[k]*S[k][j];  STh[(j-64)*64+k] = f16(S[k][j])
// ---------------------------------------------------------------------------
__global__ void __launch_bounds__(128)
grn_final_kernel(const float* __restrict__ Sfull,
                 const float* __restrict__ delta,
                 const float* __restrict__ rawLast,
                 const float* __restrict__ lastPart,
                 float* __restrict__ base,
                 _Float16* __restrict__ STh) {
  int tid = blockIdx.x * blockDim.x + threadIdx.x;
  if (tid >= NCOLS) return;
  const int j = NI + tid;
  const float invN   = 1.0f / (float)NTOT;
  const float dscale = delta[0] * invN;

  float s = 0.f;
  for (int b = 0; b < SBLK; ++b) s += lastPart[b];
  const float norm = (s > 0.f) ? 1.0f / s : 1.0f;

  float v = 0.f;
  for (int k = NI; k < NTOT; ++k)                 // rows 64..127 are zero
    v = fmaf(rawLast[k] * norm, Sfull[k * NTOT + j], v);
  base[tid] = rawLast[j] * norm + v * dscale;

  for (int k = 0; k < NI; ++k)
    STh[tid * NI + k] = (_Float16)Sfull[k * NTOT + j];
}

// ---------------------------------------------------------------------------
// f32 -> f16 convert of the input matrix (A)
// ---------------------------------------------------------------------------
__global__ void grn_conv_kernel(const float* __restrict__ in,
                                _Float16* __restrict__ Ah, int n) {
  int idx = blockIdx.x * blockDim.x + threadIdx.x;
  if (idx < n) Ah[idx] = (_Float16)in[idx];
}

// ---------------------------------------------------------------------------
// WMMA GEMM (16384x64)@(64x960), fused relu/bias epilogue, row-sum over 960
// cols, normalized write of cols 0..63. One wave per 16-row M-tile.
// ---------------------------------------------------------------------------
__global__ void __launch_bounds__(128)
grn_gemm_kernel(const _Float16* __restrict__ Ah,   // 16384x64 row-major f16
                const _Float16* __restrict__ STh,  // 960x64 row-major f16 (S^T)
                const float* __restrict__ base,    // 960
                const float* __restrict__ delta,
                float* __restrict__ out) {         // 16384x64
  const int lane  = threadIdx.x & 31;
  const int wave  = threadIdx.x >> 5;
  const int mbase = (blockIdx.x * 4 + wave) * 16;
  const int half  = lane >> 4;        // 0: lanes 0-15, 1: lanes 16-31
  const int l15   = lane & 15;
  const float dsc = delta[0] * (1.0f / (float)NTOT);

  // ---- A fragments: 16x64 as two 16x32 f16 frags (contiguous: k = 16*half + s) ----
  const _Float16* ap = Ah + (mbase + l15) * 64 + half * 16;
  v16h a0, a1;
  {
    v8h t0 = *(const v8h*)(ap);            // frag0 lo  (K 0..7  / 16..23)
    v8h t1 = *(const v8h*)(ap + 8);        // frag0 hi  (K 8..15 / 24..31)
    v8h t2 = *(const v8h*)(ap + 32);       // frag1 lo
    v8h t3 = *(const v8h*)(ap + 40);       // frag1 hi
    #pragma unroll
    for (int i = 0; i < 8; ++i) {
      a0[i] = t0[i]; a0[8 + i] = t1[i];
      a1[i] = t2[i]; a1[8 + i] = t3[i];
    }
  }

  float ps[8];
  #pragma unroll
  for (int i = 0; i < 8; ++i) ps[i] = 0.f;
  float outr[4][8];

  // B (32x16 f16): lane = column N; lanes 0-15 hold K 0..15, 16-31 hold 16..31
  auto tile = [&](int jt, float* store) {
    const int jcol = jt * 16 + l15;                    // 0..959
    const _Float16* bp = STh + jcol * 64 + half * 16;
    v8h b00 = *(const v8h*)(bp);        // frag0: K 0..31 chunk
    v8h b01 = *(const v8h*)(bp + 8);
    v8h b10 = *(const v8h*)(bp + 32);   // frag1: K 32..63 chunk
    v8h b11 = *(const v8h*)(bp + 40);
    v16h b0, b1;
    #pragma unroll
    for (int i = 0; i < 8; ++i) {
      b0[i] = b00[i]; b0[8 + i] = b01[i];
      b1[i] = b10[i]; b1[8 + i] = b11[i];
    }
    v8f acc = {};
    acc = __builtin_amdgcn_wmma_f32_16x16x32_f16(false, a0, false, b0,
                                                 (short)0, acc, false, false);
    acc = __builtin_amdgcn_wmma_f32_16x16x32_f16(false, a1, false, b1,
                                                 (short)0, acc, false, false);
    float bj = base[jcol];
    #pragma unroll
    for (int i = 0; i < 8; ++i) {
      float r = fmaxf(bj + acc[i] * dsc, 0.f);
      ps[i] += r;
      if (store) store[i] = r;
    }
  };

  #pragma unroll
  for (int jt = 0; jt < 4; ++jt) tile(jt, outr[jt]);   // cols 64..127 (output)
  for (int jt = 4; jt < 60; ++jt) tile(jt, nullptr);   // rest: sum only

  // cross-lane row-sum over the 16 lanes of each half-group
  #pragma unroll
  for (int m = 1; m < 16; m <<= 1) {
    #pragma unroll
    for (int i = 0; i < 8; ++i) ps[i] += __shfl_xor(ps[i], m, 32);
  }
  float inv[8];
  #pragma unroll
  for (int i = 0; i < 8; ++i) inv[i] = (ps[i] > 0.f) ? 1.0f / ps[i] : 1.0f;

  #pragma unroll
  for (int jt = 0; jt < 4; ++jt) {
    #pragma unroll
    for (int i = 0; i < 8; ++i) {
      int row = mbase + i + half * 8;          // C/D layout: VGPR i -> M
      out[row * 64 + jt * 16 + l15] = outr[jt][i] * inv[i];
    }
  }
}

// ---------------------------------------------------------------------------
extern "C" void kernel_launch(void* const* d_in, const int* in_sizes, int n_in,
                              void* d_out, int out_size, void* d_ws, size_t ws_size,
                              hipStream_t stream) {
  const float* inputs = (const float*)d_in[0];   // 16384*64
  const float* ids    = (const float*)d_in[1];   // 1024
  const float* enh    = (const float*)d_in[2];   // 1024
  const float* inh    = (const float*)d_in[3];   // 1024
  const float* beta   = (const float*)d_in[4];   // 1
  const float* delta  = (const float*)d_in[5];   // 1
  float* out = (float*)d_out;

  char* ws = (char*)d_ws;
  size_t off = 0;
  float*    Sfull = (float*)(ws + off);    off += (size_t)NTOT * NTOT * 4;  // 4 MB
  _Float16* Ah    = (_Float16*)(ws + off); off += (size_t)BATCH * NI * 2;   // 2 MB
  _Float16* STh   = (_Float16*)(ws + off); off += (size_t)NCOLS * NI * 2;   // 120 KB
  float*    base  = (float*)(ws + off);    off += (size_t)NCOLS * 4;
  float*    rawA  = (float*)(ws + off);    off += (size_t)NTOT * 4;
  float*    rawB  = (float*)(ws + off);    off += (size_t)NTOT * 4;
  float*    parts = (float*)(ws + off);    off += (size_t)WARMIT * SBLK * 4;

  // 1) signature matrix (1024x1024)
  grn_sig_kernel<<<(NTOT * NTOT) / 256, 256, 0, stream>>>(ids, enh, inh, beta, Sfull);
  // 2) init raw conc
  grn_init_kernel<<<1, NTOT, 0, stream>>>(rawA);
  // 3) 25 warmup steps, 30 blocks each; lazy normalization, no atomics
  float* rin = rawA;
  float* rot = rawB;
  for (int it = 0; it < WARMIT; ++it) {
    const float* pp = (it == 0) ? parts : (parts + (size_t)(it - 1) * SBLK);
    int pc = (it == 0) ? 0 : SBLK;
    grn_step_kernel<<<SBLK, 256, 0, stream>>>(Sfull, delta, rin, rot,
                                              pp, pc, parts + (size_t)it * SBLK);
    float* tmp = rin; rin = rot; rot = tmp;
  }
  // after loop, rin holds the last-written buffer
  // 4) finalize: base vector + f16 S^T slice
  grn_final_kernel<<<(NCOLS + 127) / 128, 128, 0, stream>>>(
      Sfull, delta, rin, parts + (size_t)(WARMIT - 1) * SBLK, base, STh);
  // 5) inputs -> f16
  grn_conv_kernel<<<(BATCH * NI) / 256, 256, 0, stream>>>(inputs, Ah, BATCH * NI);
  // 6) WMMA GEMM + fused epilogue (1024 M-tiles, 4 waves/block)
  grn_gemm_kernel<<<BATCH / 16 / 4, 128, 0, stream>>>(Ah, STh, base, delta, out);
}
